// BasetestLayer_84155589198303
// MI455X (gfx1250) — matrix operand order repaired and checked by reading, not compiled, not run on hardware
//
#include <hip/hip_runtime.h>
#include <hip/hip_bf16.h>
#include <math.h>

typedef __attribute__((ext_vector_type(16))) _Float16 v16h;
typedef __attribute__((ext_vector_type(8)))  _Float16 v8h;
typedef __attribute__((ext_vector_type(8)))  float    v8f;

#define N_PTS    200   // points per batch
#define NP       208   // padded to 13*16
#define NT       13    // 16x16 tiles per dim
#define D_FEAT   128
#define K_SEL    20    // greedy selections
#define N_CTX    32    // context codes
#define C_PROJ   200   // w_linear rows
#define CP       224   // proj cols padded to 7*32
#define SJ       216   // sims row stride in halves: 432B rows -> 16B aligned v8h loads
#define SIGMA_F  1.0f

__device__ __forceinline__ float fast_tanh(float x) {
  float e2 = __expf(2.0f * x);
  return 1.0f - 2.0f * __builtin_amdgcn_rcpf(e2 + 1.0f);
}

// Stage-D scratch overlays the stage-A/B X tile (dead after the Gram pass).
struct StageD {
  _Float16 MselH[32][D_FEAT];   // selected mail rows f16 (rows 20..31 zero)
  _Float16 projH[32][CP];       // tanh(proj) f16 (cols 208..223 zero)
  float    Msel[K_SEL][D_FEAT];
  float    Eh[K_SEL][D_FEAT];   // normalized cat-embedding rows
  float    svec[D_FEAT];
  float    biasArr[K_SEL];
  float    Wts[32][33];         // weights + bias (pad 33 vs conflicts)
  float    Wsm[32][21];         // softmaxed (K_ctx x L)
  float    rnArr[K_SEL];
  int      catSel[K_SEL];
};
struct StageA {
  _Float16 Xh[NP][D_FEAT];      // mail[b] in f16, padded rows zero
};
union ShMem { StageA a; StageD d; };

__global__ __launch_bounds__(256, 1) void fused_poly_kernel(
    const float* __restrict__ mail, const int* __restrict__ cat,
    const float* __restrict__ cemb, const float* __restrict__ wlin,
    const float* __restrict__ ccodes, float* __restrict__ out)
{
  __shared__ __align__(64) ShMem u;
  __shared__ __align__(64) _Float16 simsH[NP * SJ];
  __shared__ float sq[NP];
  __shared__ float cache[NP];
  __shared__ float gainArr[256];
  __shared__ float redArr[8];
  __shared__ int   selIdx[K_SEL];
  __shared__ float scaleSh;
  __shared__ int   selSh;

  const int b    = blockIdx.x;
  const int tid  = threadIdx.x;
  const int wid  = tid >> 5;
  const int lane = tid & 31;
  const int m    = lane & 15;   // WMMA column/row lane index
  const int hi   = lane >> 4;   // lane half selector
  const size_t mbase = (size_t)b * (N_PTS * D_FEAT);

  // ---------------- Stage A: stage mail[b] into LDS (f16) + f32 row norms
  for (int idx = tid; idx < N_PTS * D_FEAT; idx += 256)
    u.a.Xh[idx >> 7][idx & 127] = (_Float16)mail[mbase + idx];
  for (int idx = tid; idx < (NP - N_PTS) * D_FEAT; idx += 256)
    u.a.Xh[N_PTS + (idx >> 7)][idx & 127] = (_Float16)0.0f;
  // Warm L2/WGP$ with the stage-D weight matrices (needed ~after greedy).
  for (int p = tid; p < 800; p += 256)       // 200*128 f32 = 800 x 128B lines
    __builtin_prefetch(wlin + p * 32, 0, 0);
  for (int p = tid; p < 200; p += 256)       // 32*200 f32 = 200 x 128B lines
    __builtin_prefetch(ccodes + p * 32, 0, 0);
  if (tid < NP) {
    float s = 0.0f;
    if (tid < N_PTS) {
      const float* row = mail + mbase + (size_t)tid * D_FEAT;
      #pragma unroll 4
      for (int d = 0; d < D_FEAT; ++d) s += row[d] * row[d];
    }
    sq[tid] = s;
    cache[tid] = 0.0f;
  }
  __syncthreads();

  // ---------------- Stage B: Gram via v_wmma_f32_16x16x32_f16 -> distances
  // Each wave owns whole tile rows: A fragments loaded once, reused 13x.
  float dsum = 0.0f;
  for (int ti = wid; ti < NT; ti += 8) {
    const int rowA = ti * 16 + m;
    v16h Afr[4];
    #pragma unroll
    for (int kc = 0; kc < 4; ++kc) {
      const int k0 = kc * 32;
      v8h a0 = *(const v8h*)&u.a.Xh[rowA][k0 + 16 * hi];
      v8h a1 = *(const v8h*)&u.a.Xh[rowA][k0 + 8 + 16 * hi];
      #pragma unroll
      for (int i = 0; i < 8; ++i) { Afr[kc][i] = a0[i]; Afr[kc][i + 8] = a1[i]; }
    }
    for (int tj = 0; tj < NT; ++tj) {
      const int rowB = tj * 16 + m;
      v8f acc = {};
      #pragma unroll
      for (int kc = 0; kc < 4; ++kc) {
        v16h Bv = *(const v16h*)&u.a.Xh[rowB][kc * 32 + 16 * hi];
        acc = __builtin_amdgcn_wmma_f32_16x16x32_f16(false, Afr[kc], false, Bv,
                                                     (short)0, acc, false, false);
      }
      #pragma unroll
      for (int v = 0; v < 8; ++v) {
        const int i = ti * 16 + v + 8 * hi;
        const int j = tj * 16 + m;
        float d2 = sq[i] + sq[j] - 2.0f * acc[v];
        float dd = __builtin_amdgcn_sqrtf(fmaxf(d2, 0.0f));
        simsH[i * SJ + j] = (_Float16)dd;
        if (i < N_PTS && j < N_PTS) dsum += dd;
      }
    }
  }
  #pragma unroll
  for (int off = 16; off; off >>= 1) dsum += __shfl_xor(dsum, off, 32);
  if (lane == 0) redArr[wid] = dsum;
  __syncthreads();
  if (tid == 0) {
    float tot = 0.0f;
    for (int w = 0; w < 8; ++w) tot += redArr[w];
    scaleSh = SIGMA_F * tot * (1.0f / (float)(N_PTS * N_PTS));
  }
  __syncthreads();
  const float negInv = -1.0f / scaleSh;
  for (int idx = tid; idx < NP * NP; idx += 256) {
    const int i = idx / NP, j = idx % NP;
    float s = 0.0f;
    if (i < N_PTS && j < N_PTS)
      s = __expf((float)simsH[i * SJ + j] * negInv);
    simsH[i * SJ + j] = (_Float16)s;
  }
  __syncthreads();

  // ---------------- Stage C: greedy facility-location (k=20)
  for (int it = 0; it < K_SEL; ++it) {
    float gain = -1.0f;
    if (tid < N_PTS) {
      gain = 0.0f;
      const _Float16* srow = &simsH[tid * SJ];
      for (int j = 0; j < N_PTS; j += 8) {       // 16B-aligned v8h loads
        v8h sv = *(const v8h*)(srow + j);
        #pragma unroll
        for (int q = 0; q < 8; ++q) {
          float s = (float)sv[q];
          float c = cache[j + q];
          gain += fmaxf(s, c) - c;
        }
      }
    }
    gainArr[tid] = gain;
    __syncthreads();
    if (tid == 0) {                       // first-max-on-tie, matches jnp.argmax
      float best = gainArr[0]; int bi = 0;
      for (int i2 = 1; i2 < N_PTS; ++i2) {
        float g = gainArr[i2];
        if (g > best) { best = g; bi = i2; }
      }
      selIdx[it] = bi; selSh = bi;
    }
    __syncthreads();
    const int sel = selSh;
    if (tid < N_PTS)
      cache[tid] = fmaxf(cache[tid], (float)simsH[sel * SJ + tid]);
    __syncthreads();
  }

  // ---------------- Stage D: gather + cosine bias + poly attention
  if (tid < K_SEL) u.d.catSel[tid] = cat[(size_t)b * N_PTS + selIdx[tid]];
  for (int idx = tid; idx < K_SEL * D_FEAT; idx += 256) {
    const int r = idx >> 7, d = idx & 127;
    float v = mail[mbase + (size_t)selIdx[r] * D_FEAT + d];
    u.d.Msel[r][d]  = v;
    u.d.MselH[r][d] = (_Float16)v;
  }
  for (int idx = tid; idx < (32 - K_SEL) * D_FEAT; idx += 256)
    u.d.MselH[K_SEL + (idx >> 7)][idx & 127] = (_Float16)0.0f;
  for (int idx = tid; idx < 32 * (CP - 208); idx += 256)
    u.d.projH[idx >> 4][208 + (idx & 15)] = (_Float16)0.0f;
  __syncthreads();
  for (int idx = tid; idx < K_SEL * D_FEAT; idx += 256) {
    const int r = idx >> 7, d = idx & 127;
    u.d.Eh[r][d] = cemb[(size_t)u.d.catSel[r] * D_FEAT + d];
  }
  __syncthreads();
  if (tid < K_SEL) {
    float s = 0.0f;
    for (int d = 0; d < D_FEAT; ++d) { float v = u.d.Eh[tid][d]; s += v * v; }
    u.d.rnArr[tid] = __builtin_amdgcn_rsqf(s);
  }
  __syncthreads();
  for (int idx = tid; idx < K_SEL * D_FEAT; idx += 256)
    u.d.Eh[idx >> 7][idx & 127] *= u.d.rnArr[idx >> 7];
  __syncthreads();
  if (tid < D_FEAT) {
    float s = 0.0f;
    for (int r = 0; r < K_SEL; ++r) s += u.d.Eh[r][tid];
    u.d.svec[tid] = s;
  }
  __syncthreads();
  // bias[l] = mean_j cos(l,j) = (e_hat_l . sum_j e_hat_j) / 20
  if (tid < K_SEL) {
    float s = 0.0f;
    for (int d = 0; d < D_FEAT; ++d) s += u.d.Eh[tid][d] * u.d.svec[d];
    u.d.biasArr[tid] = s * (1.0f / (float)K_SEL);
  }
  __syncthreads();

  // proj = tanh(Msel @ W^T): (32 x 208) via WMMA, k = 128
  for (int t = wid; t < 2 * NT; t += 8) {
    const int mt = t / NT, ct = t % NT;
    v8f acc = {};
    #pragma unroll
    for (int kc = 0; kc < 4; ++kc) {
      const int k0 = kc * 32;
      const int rowA = mt * 16 + m;
      v8h a0 = *(const v8h*)&u.d.MselH[rowA][k0 + 16 * hi];
      v8h a1 = *(const v8h*)&u.d.MselH[rowA][k0 + 8 + 16 * hi];
      v16h A;
      #pragma unroll
      for (int i = 0; i < 8; ++i) { A[i] = a0[i]; A[i + 8] = a1[i]; }
      const int rowW = ct * 16 + m;
      v16h Bv;
      if (rowW < C_PROJ) {
        const float* wp = wlin + (size_t)rowW * D_FEAT + k0 + 16 * hi;
        #pragma unroll
        for (int i = 0; i < 16; ++i) Bv[i] = (_Float16)wp[i];
      } else {
        #pragma unroll
        for (int i = 0; i < 16; ++i) Bv[i] = (_Float16)0.0f;
      }
      acc = __builtin_amdgcn_wmma_f32_16x16x32_f16(false, A, false, Bv,
                                                   (short)0, acc, false, false);
    }
    #pragma unroll
    for (int v = 0; v < 8; ++v) {
      const int l = mt * 16 + v + 8 * hi;
      const int c = ct * 16 + m;
      u.d.projH[l][c] = (_Float16)fast_tanh(acc[v]);
    }
  }
  __syncthreads();

  // weights = projH @ CC^T: (32 x 32) via WMMA, k padded to 224
  if (wid < 4) {
    const int mt = wid >> 1, nt = wid & 1;
    v8f acc = {};
    for (int kc = 0; kc < 7; ++kc) {
      const int k0 = kc * 32;
      const int rowA = mt * 16 + m;
      v8h a0 = *(const v8h*)&u.d.projH[rowA][k0 + 16 * hi];
      v8h a1 = *(const v8h*)&u.d.projH[rowA][k0 + 8 + 16 * hi];
      v16h A;
      #pragma unroll
      for (int i = 0; i < 8; ++i) { A[i] = a0[i]; A[i + 8] = a1[i]; }
      const int rowC = nt * 16 + m;
      v16h Bv;
      #pragma unroll
      for (int i = 0; i < 16; ++i) {
        const int cidx = k0 + 16 * hi + i;
        Bv[i] = (cidx < C_PROJ) ? (_Float16)ccodes[rowC * C_PROJ + cidx]
                                : (_Float16)0.0f;
      }
      acc = __builtin_amdgcn_wmma_f32_16x16x32_f16(false, A, false, Bv,
                                                   (short)0, acc, false, false);
    }
    #pragma unroll
    for (int v = 0; v < 8; ++v) {
      const int l  = mt * 16 + v + 8 * hi;
      const int kx = nt * 16 + m;
      float val = acc[v];
      if (l < K_SEL) val += u.d.biasArr[l];
      u.d.Wts[l][kx] = val;
    }
  }
  __syncthreads();

  // softmax over L=20 per context code
  if (tid < N_CTX) {
    float mx = -1e30f;
    for (int l = 0; l < K_SEL; ++l) mx = fmaxf(mx, u.d.Wts[l][tid]);
    float s = 0.0f;
    for (int l = 0; l < K_SEL; ++l) {
      float e = __expf(u.d.Wts[l][tid] - mx);
      u.d.Wsm[tid][l] = e;
      s += e;
    }
    float inv = __builtin_amdgcn_rcpf(s);
    for (int l = 0; l < K_SEL; ++l) u.d.Wsm[tid][l] *= inv;
  }
  __syncthreads();

  // out = Wsm (32x20) @ Msel (20x128)
  for (int o = tid; o < N_CTX * D_FEAT; o += 256) {
    const int kx = o >> 7, d = o & 127;
    float acc = 0.0f;
    #pragma unroll
    for (int l = 0; l < K_SEL; ++l) acc += u.d.Wsm[kx][l] * u.d.Msel[l][d];
    out[(size_t)b * (N_CTX * D_FEAT) + o] = acc;
  }
}

extern "C" void kernel_launch(void* const* d_in, const int* in_sizes, int n_in,
                              void* d_out, int out_size, void* d_ws, size_t ws_size,
                              hipStream_t stream) {
  const float* mail   = (const float*)d_in[0];
  const int*   cat    = (const int*)d_in[1];
  const float* cemb   = (const float*)d_in[2];
  const float* wlin   = (const float*)d_in[3];
  const float* ccodes = (const float*)d_in[4];
  float* out = (float*)d_out;
  const int B = in_sizes[0] / (N_PTS * D_FEAT);
  hipLaunchKernelGGL(fused_poly_kernel, dim3(B), dim3(256), 0, stream,
                     mail, cat, cemb, wlin, ccodes, out);
}
